// MultiHeadedAttention_4561255268493
// MI455X (gfx1250) — compile-verified, matrix operand order checked by reading, not run on hardware
//
#include <hip/hip_runtime.h>

// Problem constants (from reference)
#define BSZ   8
#define LEN   512
#define HDIM  1024
#define NHEAD 16
#define PN    256
#define DKD   64

typedef __attribute__((ext_vector_type(16))) __bf16 v16bf;
typedef __attribute__((ext_vector_type(8)))  __bf16 v8bf;
typedef __attribute__((ext_vector_type(4)))  __bf16 v4bf;
typedef __attribute__((ext_vector_type(8)))  float  v8f;

// float -> bf16, round-to-nearest-even
static __device__ __forceinline__ __bf16 f2bf(float f) {
  unsigned u = __builtin_bit_cast(unsigned, f);
  unsigned r = u + 0x7fffu + ((u >> 16) & 1u);
  unsigned short h = (unsigned short)(r >> 16);
  return __builtin_bit_cast(__bf16, h);
}

// ISA 7.12.2 bf16 A(16x32)/B(32x16) fragment: lanes 0-15 hold row=lane with
// K = {0..7, 16..23}; lanes 16-31 hold row=lane-16 with K = {8..15, 24..31}.
// With bf16 row-major storage this is exactly two contiguous 16-byte loads.
static __device__ __forceinline__ v16bf load_nk_bf(const __bf16* base, int ld,
                                                   int lane) {
  int row = lane & 15;
  int kb = (lane & 16) ? 8 : 0;
  const __bf16* p = base + row * ld + kb;
  v8bf lo = *(const v8bf*)p;         // K = kb .. kb+7
  v8bf hi = *(const v8bf*)(p + 16);  // K = kb+16 .. kb+23
  return __builtin_shufflevector(lo, hi, 0, 1, 2, 3, 4, 5, 6, 7, 8, 9, 10, 11,
                                 12, 13, 14, 15);
}

static __device__ __forceinline__ v8f wmma_bf(v16bf a, v16bf b, v8f c) {
  return __builtin_amdgcn_wmma_f32_16x16x32_bf16(false, a, false, b, (short)0, c,
                                                 false, false);
}

// ---------------- conversion kernels ----------------

// elementwise f32 -> bf16, 8 elements/thread, 16B vector stores
__global__ void __launch_bounds__(256)
cvt_f32_bf16(const float* __restrict__ src, __bf16* __restrict__ dst, long n) {
  long i = ((long)blockIdx.x * blockDim.x + threadIdx.x) * 8;
  if (i >= n) return;
  float4 f0 = *(const float4*)(src + i);
  float4 f1 = *(const float4*)(src + i + 4);
  v8bf o;
  o[0] = f2bf(f0.x); o[1] = f2bf(f0.y); o[2] = f2bf(f0.z); o[3] = f2bf(f0.w);
  o[4] = f2bf(f1.x); o[5] = f2bf(f1.y); o[6] = f2bf(f1.z); o[7] = f2bf(f1.w);
  *(v8bf*)(dst + i) = o;
}

// r_v (b,h,p,d) f32 -> rvT (b,h,d,p) bf16
__global__ void __launch_bounds__(256)
tconv_rv(const float* __restrict__ rv, __bf16* __restrict__ rvT) {
  long bh = blockIdx.x;
  const float* s = rv + bh * PN * DKD;
  __bf16* d = rvT + bh * PN * DKD;
  for (int t = threadIdx.x; t < PN * DKD; t += 256) {
    int p = t / DKD, k = t % DKD;
    d[(long)k * PN + p] = f2bf(s[t]);
  }
}

// ---------------- GEMM: C = A(MxK,bf16) @ W(NxK,bf16)^T + bias ----------------
// MODE 0: f32 row-major out; MODE 1: bf16 row-major out;
// MODE 2: bf16 out pre-transposed per head: vT[((b*NH+h)*DK+d)*LEN + l]
template <int MODE>
__global__ void __launch_bounds__(32)
gemm_bf16(const __bf16* __restrict__ A, const __bf16* __restrict__ W,
          const float* __restrict__ bias, void* __restrict__ Cout, int M, int N,
          int K) {
  int lane = threadIdx.x;
  int nTiles = N >> 6;
  int m0 = (blockIdx.x / nTiles) << 4;
  int n0 = (blockIdx.x % nTiles) << 6;

  v8f acc[4] = {};
  for (int k0 = 0; k0 < K; k0 += 32) {
    __builtin_prefetch((const void*)(A + (long)m0 * K + k0 + 256), 0, 0);
    v16bf a = load_nk_bf(A + (long)m0 * K + k0, K, lane);
#pragma unroll
    for (int nt = 0; nt < 4; ++nt) {
      v16bf b = load_nk_bf(W + (long)(n0 + nt * 16) * K + k0, K, lane);
      acc[nt] = wmma_bf(a, b, acc[nt]);
    }
  }
  int nL = lane & 15;
  int mo = (lane & 16) ? 8 : 0;
#pragma unroll
  for (int nt = 0; nt < 4; ++nt) {
    int n = n0 + nt * 16 + nL;
    float bv = bias[n];
#pragma unroll
    for (int r = 0; r < 8; ++r) {
      int m = m0 + mo + r;
      float val = acc[nt][r] + bv;
      if constexpr (MODE == 0) {
        ((float*)Cout)[(long)m * N + n] = val;
      } else if constexpr (MODE == 1) {
        ((__bf16*)Cout)[(long)m * N + n] = f2bf(val);
      } else {
        int b = m >> 9, l = m & (LEN - 1);   // LEN = 512
        int h = n >> 6, d = n & (DKD - 1);   // DKD = 64
        ((__bf16*)Cout)[((long)((b * NHEAD + h) * DKD + d)) * LEN + l] =
            f2bf(val);
      }
    }
  }
}

// ---------------- attention: one wave per (b, h, 16-row query tile) ----------
__global__ void __launch_bounds__(32)
attn_kernel(const __bf16* __restrict__ qp, const __bf16* __restrict__ kp,
            const __bf16* __restrict__ vT, const int* __restrict__ mask,
            const __bf16* __restrict__ rkb, const __bf16* __restrict__ rvT,
            const int* __restrict__ path_map, const float* __restrict__ ap,
            __bf16* __restrict__ xo) {
  __shared__ __align__(16) float  s_p[16 * LEN];   // scores / exp
  __shared__ __align__(16) float  s_aux[16 * PN];  // rel, then f32 scatter
  __shared__ __align__(16) __bf16 s_pb[16 * LEN];  // probabilities (bf16)
  __shared__ __align__(16) __bf16 s_ab[16 * PN];   // scatter buckets (bf16)

  int lane = threadIdx.x;
  int tilesPerBH = LEN / 16;
  int it = blockIdx.x % tilesPerBH;
  int bh = blockIdx.x / tilesPerBH;
  int h = bh % NHEAD;
  int b = bh / NHEAD;
  int i0 = it * 16;
  int nL = lane & 15;
  int mo = (lane & 16) ? 8 : 0;

  const int*   pmrow = path_map + (long)(b * LEN + i0) * LEN;
  const float* aprow = ap + (long)((b * NHEAD + h) * LEN + i0) * LEN;
  const int*   mrow  = mask + (long)b * LEN * LEN + (long)i0 * LEN;

  // Q tile fragments (16x64 -> two 16x32 bf16 frags)
  const __bf16* qrow = qp + (long)(b * LEN + i0) * HDIM + h * DKD;
  v16bf aq0 = load_nk_bf(qrow, HDIM, lane);
  v16bf aq1 = load_nk_bf(qrow + 32, HDIM, lane);

  // rel = q @ r_k^T (16 x PN) -> s_aux; groups of 4 tiles, grouped loads
  const __bf16* rk0 = rkb + (long)(b * NHEAD + h) * PN * DKD;
  for (int g = 0; g < PN / 64; ++g) {
    v16bf rb[8];
#pragma unroll
    for (int t = 0; t < 4; ++t) {
      const __bf16* rk = rk0 + (g * 4 + t) * 16 * DKD;
      rb[2 * t]     = load_nk_bf(rk, DKD, lane);
      rb[2 * t + 1] = load_nk_bf(rk + 32, DKD, lane);
    }
#pragma unroll
    for (int t = 0; t < 4; ++t) {
      v8f acc = {};
      acc = wmma_bf(aq0, rb[2 * t], acc);
      acc = wmma_bf(aq1, rb[2 * t + 1], acc);
#pragma unroll
      for (int r = 0; r < 8; ++r)
        s_aux[(mo + r) * PN + (g * 4 + t) * 16 + nL] = acc[r];
    }
  }
  __syncthreads();

  // Phase A: raw scores = q k^T -> s_p
  // groups of 4 tiles (8 b128-pair loads per group) + cross-group double buffer
  {
    const __bf16* kbase = kp + (long)(b * LEN) * HDIM + h * DKD;
    v16bf kb[8], nb[8];
#pragma unroll
    for (int t = 0; t < 4; ++t) {
      const __bf16* p = kbase + (long)(t * 16) * HDIM;
      kb[2 * t]     = load_nk_bf(p, HDIM, lane);
      kb[2 * t + 1] = load_nk_bf(p + 32, HDIM, lane);
    }
    for (int g = 0; g < LEN / 64; ++g) {
#pragma unroll
      for (int t = 0; t < 8; ++t) nb[t] = kb[t];
      if (g + 1 < LEN / 64) {
#pragma unroll
        for (int t = 0; t < 4; ++t) {
          const __bf16* p = kbase + (long)((g + 1) * 64 + t * 16) * HDIM;
          nb[2 * t]     = load_nk_bf(p, HDIM, lane);
          nb[2 * t + 1] = load_nk_bf(p + 32, HDIM, lane);
        }
      }
#pragma unroll
      for (int t = 0; t < 4; ++t) {
        v8f acc = {};
        acc = wmma_bf(aq0, kb[2 * t], acc);
        acc = wmma_bf(aq1, kb[2 * t + 1], acc);
        int j = (g * 4 + t) * 16 + nL;
#pragma unroll
        for (int r = 0; r < 8; ++r)
          s_p[(mo + r) * LEN + j] = acc[r];
      }
#pragma unroll
      for (int t = 0; t < 8; ++t) kb[t] = nb[t];
    }
  }
  __syncthreads();

  // Phase B: vectorized bias/mask: s = (qk + rel[pm] + ap)/8, masked
  // (path_map read as int4 straight from global; L2-resident, shared by heads)
  for (int c = 0; c < (16 * LEN / 4) / 32; ++c) {
    int t4 = c * 32 + lane;
    int m = t4 >> 7;
    int j = (t4 & 127) << 2;
    float4 sc  = *(float4*)(s_p + m * LEN + j);
    float4 apv = *(const float4*)(aprow + m * LEN + j);
    int4   mk  = *(const int4*)(mrow + m * LEN + j);
    int4   pm4 = *(const int4*)(pmrow + m * LEN + j);
    float r0 = s_aux[m * PN + pm4.x];
    float r1 = s_aux[m * PN + pm4.y];
    float r2 = s_aux[m * PN + pm4.z];
    float r3 = s_aux[m * PN + pm4.w];
    sc.x = (mk.x == 0) ? -1.0e9f : (sc.x + r0 + apv.x) * 0.125f;
    sc.y = (mk.y == 0) ? -1.0e9f : (sc.y + r1 + apv.y) * 0.125f;
    sc.z = (mk.z == 0) ? -1.0e9f : (sc.z + r2 + apv.z) * 0.125f;
    sc.w = (mk.w == 0) ? -1.0e9f : (sc.w + r3 + apv.w) * 0.125f;
    *(float4*)(s_p + m * LEN + j) = sc;
  }
  __syncthreads();

  // reset f32 scatter buckets
  for (int t = lane; t < 16 * PN; t += 32) s_aux[t] = 0.0f;
  __syncthreads();

  // Phase C: softmax, two lanes per row (each owns 256 columns)
  {
    int m = lane & 15;
    int j0 = (lane >> 4) * (LEN / 2);
    float* row = s_p + m * LEN + j0;
    float mx = -3.0e38f;
    for (int jc = 0; jc < LEN / 2 / 4; ++jc) {
      float4 v = *(float4*)(row + jc * 4);
      mx = fmaxf(mx, fmaxf(fmaxf(v.x, v.y), fmaxf(v.z, v.w)));
    }
    mx = fmaxf(mx, __shfl_xor(mx, 16, 32));
    float sum = 0.0f;
    for (int jc = 0; jc < LEN / 2 / 4; ++jc) {
      float4 v = *(float4*)(row + jc * 4);
      v.x = __expf(v.x - mx); v.y = __expf(v.y - mx);
      v.z = __expf(v.z - mx); v.w = __expf(v.w - mx);
      *(float4*)(row + jc * 4) = v;
      sum += (v.x + v.y) + (v.z + v.w);
    }
    sum += __shfl_xor(sum, 16, 32);
    float inv = 1.0f / sum;
    for (int jc = 0; jc < LEN / 2 / 4; ++jc) {
      int j = j0 + jc * 4;
      float4 v = *(float4*)(s_p + m * LEN + j);
      int4 pm4 = *(const int4*)(pmrow + m * LEN + j);
      float p0 = v.x * inv, p1 = v.y * inv, p2 = v.z * inv, p3 = v.w * inv;
      v4bf pb; pb[0] = f2bf(p0); pb[1] = f2bf(p1); pb[2] = f2bf(p2); pb[3] = f2bf(p3);
      *(v4bf*)(s_pb + m * LEN + j) = pb;
      atomicAdd(&s_aux[m * PN + pm4.x], p0);
      atomicAdd(&s_aux[m * PN + pm4.y], p1);
      atomicAdd(&s_aux[m * PN + pm4.z], p2);
      atomicAdd(&s_aux[m * PN + pm4.w], p3);
    }
  }
  __syncthreads();
  // scatter buckets -> bf16
  for (int t = lane; t < 16 * PN; t += 32) s_ab[t] = f2bf(s_aux[t]);
  __syncthreads();

  // Phase D: x = P @ V + scatter @ r_v (16 x 64); double-buffered B frags
  v8f xacc[4] = {};
  {
    const __bf16* vbase = vT + (long)((b * NHEAD + h) * DKD) * LEN;
    v16bf vb[4], nvb[4];
#pragma unroll
    for (int nt = 0; nt < 4; ++nt)
      vb[nt] = load_nk_bf(vbase + (long)(nt * 16) * LEN, LEN, lane);
    for (int kk = 0; kk < LEN / 32; ++kk) {
      v16bf a = load_nk_bf(s_pb + kk * 32, LEN, lane);
#pragma unroll
      for (int nt = 0; nt < 4; ++nt) nvb[nt] = vb[nt];
      if (kk + 1 < LEN / 32) {
#pragma unroll
        for (int nt = 0; nt < 4; ++nt)
          nvb[nt] =
              load_nk_bf(vbase + (long)(nt * 16) * LEN + (kk + 1) * 32, LEN, lane);
      }
#pragma unroll
      for (int nt = 0; nt < 4; ++nt) xacc[nt] = wmma_bf(a, vb[nt], xacc[nt]);
#pragma unroll
      for (int nt = 0; nt < 4; ++nt) vb[nt] = nvb[nt];
    }
  }
  {
    const __bf16* rvbase = rvT + (long)((b * NHEAD + h) * DKD) * PN;
    v16bf rb[4], nrb[4];
#pragma unroll
    for (int nt = 0; nt < 4; ++nt)
      rb[nt] = load_nk_bf(rvbase + (long)(nt * 16) * PN, PN, lane);
    for (int kk = 0; kk < PN / 32; ++kk) {
      v16bf a = load_nk_bf(s_ab + kk * 32, PN, lane);
#pragma unroll
      for (int nt = 0; nt < 4; ++nt) nrb[nt] = rb[nt];
      if (kk + 1 < PN / 32) {
#pragma unroll
        for (int nt = 0; nt < 4; ++nt)
          nrb[nt] =
              load_nk_bf(rvbase + (long)(nt * 16) * PN + (kk + 1) * 32, PN, lane);
      }
#pragma unroll
      for (int nt = 0; nt < 4; ++nt) xacc[nt] = wmma_bf(a, rb[nt], xacc[nt]);
#pragma unroll
      for (int nt = 0; nt < 4; ++nt) rb[nt] = nrb[nt];
    }
  }
#pragma unroll
  for (int nt = 0; nt < 4; ++nt)
#pragma unroll
    for (int r = 0; r < 8; ++r)
      xo[(long)(b * LEN + i0 + mo + r) * HDIM + h * DKD + nt * 16 + nL] =
          f2bf(xacc[nt][r]);
}

// ---------------- host side ----------------

extern "C" void kernel_launch(void* const* d_in, const int* in_sizes, int n_in,
                              void* d_out, int out_size, void* d_ws,
                              size_t ws_size, hipStream_t stream) {
  (void)in_sizes; (void)n_in; (void)out_size; (void)ws_size;
  const float* q_in = (const float*)d_in[0];
  const float* k_in = (const float*)d_in[1];
  const float* v_in = (const float*)d_in[2];
  const int*   mask = (const int*)d_in[3];
  const float* r_k  = (const float*)d_in[4];
  const float* r_v  = (const float*)d_in[5];
  const int*   path_map = (const int*)d_in[6];
  const float* ap   = (const float*)d_in[7];
  const float* Wq = (const float*)d_in[8];  const float* bq = (const float*)d_in[9];
  const float* Wk = (const float*)d_in[10]; const float* bk = (const float*)d_in[11];
  const float* Wv = (const float*)d_in[12]; const float* bv = (const float*)d_in[13];
  const float* Wo = (const float*)d_in[14]; const float* bo = (const float*)d_in[15];
  float* out = (float*)d_out;

  const long S = (long)BSZ * LEN * HDIM;   // 4,194,304 activation elements
  const long WSZ = (long)HDIM * HDIM;      // 1,048,576 weight elements
  const long RSZ = (long)BSZ * NHEAD * PN * DKD;  // 2,097,152 relation elems

  __bf16* a_tmp = (__bf16*)d_ws;           // staged GEMM A operand (reused)
  __bf16* w0 = a_tmp + S;                  // Wq
  __bf16* w1 = w0 + WSZ;                   // Wk
  __bf16* w2 = w1 + WSZ;                   // Wv
  __bf16* w3 = w2 + WSZ;                   // Wo
  __bf16* qp_b = w3 + WSZ;                 // Q projection (bf16)
  __bf16* kp_b = qp_b + S;                 // K projection (bf16)
  __bf16* vTb  = kp_b + S;                 // V projection, per-head transposed
  __bf16* rkb  = vTb + S;                  // r_k bf16
  __bf16* rvT  = rkb + RSZ;                // r_v transposed bf16
  __bf16* xo_b = rvT + RSZ;                // attention output bf16

  const int M = BSZ * LEN, N = HDIM, K = HDIM;
  dim3 b32(32), b256(256);
  int gemmGrid = (M / 16) * (N / 64);
  int gS = (int)(S / (8 * 256));
  int gW = (int)(WSZ / (8 * 256));
  int gR = (int)(RSZ / (8 * 256));

  // weight / relation conversions
  cvt_f32_bf16<<<gW, b256, 0, stream>>>(Wq, w0, WSZ);
  cvt_f32_bf16<<<gW, b256, 0, stream>>>(Wk, w1, WSZ);
  cvt_f32_bf16<<<gW, b256, 0, stream>>>(Wv, w2, WSZ);
  cvt_f32_bf16<<<gW, b256, 0, stream>>>(Wo, w3, WSZ);
  cvt_f32_bf16<<<gR, b256, 0, stream>>>(r_k, rkb, RSZ);
  tconv_rv<<<BSZ * NHEAD, b256, 0, stream>>>(r_v, rvT);

  // projections (bf16 outputs; V stored per-head transposed)
  cvt_f32_bf16<<<gS, b256, 0, stream>>>(q_in, a_tmp, S);
  gemm_bf16<1><<<gemmGrid, b32, 0, stream>>>(a_tmp, w0, bq, qp_b, M, N, K);
  cvt_f32_bf16<<<gS, b256, 0, stream>>>(k_in, a_tmp, S);
  gemm_bf16<1><<<gemmGrid, b32, 0, stream>>>(a_tmp, w1, bk, kp_b, M, N, K);
  cvt_f32_bf16<<<gS, b256, 0, stream>>>(v_in, a_tmp, S);
  gemm_bf16<2><<<gemmGrid, b32, 0, stream>>>(a_tmp, w2, bv, vTb, M, N, K);

  // fused relation-aware attention
  int attnGrid = BSZ * NHEAD * (LEN / 16);
  attn_kernel<<<attnGrid, b32, 0, stream>>>(qp_b, kp_b, vTb, mask, rkb, rvT,
                                            path_map, ap, xo_b);

  // output projection (f32 out)
  gemm_bf16<0><<<gemmGrid, b32, 0, stream>>>(xo_b, w3, bo, out, M, N, K);
}
